// HeteroGNNBlock_40492951666849
// MI455X (gfx1250) — matrix operand order repaired and checked by reading, not compiled, not run on hardware
//
#include <hip/hip_runtime.h>

// ---------------------------------------------------------------------------
// HeteroGNN block for MI455X (gfx1250, wave32, WMMA).
// Dense math done in bf16 WMMA (v_wmma_f32_16x16x32_bf16), f32 accumulate.
// Hidden (B,N,256) intermediate is fused away (would be 205MB > 192MB L2).
// ---------------------------------------------------------------------------

typedef __attribute__((ext_vector_type(16))) __bf16 v16bf;
typedef __attribute__((ext_vector_type(8)))  __bf16 v8bf;
typedef __attribute__((ext_vector_type(8)))  float  v8f;

#define D_IN   128
#define H_HID  256
#define K_CAT  384          // [aggA | x | aggB]
#define MT     32           // rows per workgroup in fused kernel
#define LN_EPSF 1e-5f

__device__ __forceinline__ __bf16 f2bf(float x) {
  unsigned u = __builtin_bit_cast(unsigned, x);
  u += 0x7FFFu + ((u >> 16) & 1u);           // round-to-nearest-even
  unsigned short h = (unsigned short)(u >> 16);
  return __builtin_bit_cast(__bf16, h);
}

__device__ __forceinline__ v16bf frag_pair(const __bf16* p0, const __bf16* p1) {
  v8bf lo = *(const v8bf*)p0;
  v8bf hi = *(const v8bf*)p1;
  return __builtin_shufflevector(lo, hi, 0,1,2,3,4,5,6,7,8,9,10,11,12,13,14,15);
}

// ---------------------------------------------------------------------------
__global__ void zero_f32_kernel(float* __restrict__ p, long n) {
  long i = (long)blockIdx.x * blockDim.x + threadIdx.x;
  long stride = (long)gridDim.x * blockDim.x;
  for (; i < n; i += stride) p[i] = 0.0f;
}

// One thread per (edge, feature). Handles both batches. cnt is batch-invariant.
__global__ void scatter_accum_kernel(const float* __restrict__ x_src,
                                     const int* __restrict__ edge,   // [2,E]: src row, dst row
                                     float* __restrict__ agg,        // (B, n_dst, 128)
                                     float* __restrict__ cnt,        // (n_dst)
                                     int E, int n_src, int n_dst) {
  int idx = blockIdx.x * blockDim.x + threadIdx.x;
  int total = E * D_IN;
  if (idx >= total) return;
  int e = idx >> 7;
  int d = idx & (D_IN - 1);
  int s = edge[e];
  int t = edge[E + e];
  float v0 = x_src[(size_t)s * D_IN + d];
  float v1 = x_src[((size_t)n_src + s) * D_IN + d];
  atomicAdd(agg + (size_t)t * D_IN + d, v0);
  atomicAdd(agg + ((size_t)n_dst + t) * D_IN + d, v1);
  if (d == 0) atomicAdd(cnt + t, 1.0f);
}

// Build Xcat (B*N, 384) bf16 = [aggA/max(cntA,1) | z | aggB/max(cntB,1)]
__global__ void pack_xcat_kernel(const float* __restrict__ z,
                                 const float* __restrict__ aggA,
                                 const float* __restrict__ aggB,
                                 const float* __restrict__ cntA,
                                 const float* __restrict__ cntB,
                                 __bf16* __restrict__ xcat, int N) {
  int idx = blockIdx.x * blockDim.x + threadIdx.x;
  int total = 2 * N * D_IN;
  if (idx >= total) return;
  int r = idx >> 7;              // b*N + n
  int d = idx & (D_IN - 1);
  int n = r % N;
  float cA = fmaxf(cntA[n], 1.0f);
  float cB = fmaxf(cntB[n], 1.0f);
  size_t base = (size_t)r * K_CAT;
  xcat[base + d]           = f2bf(aggA[idx] / cA);
  xcat[base + D_IN + d]    = f2bf(z[idx]);
  xcat[base + 2*D_IN + d]  = f2bf(aggB[idx] / cB);
}

// W1 (256,384) bf16 = [WlA | WrA+WrB | WlB]; b1 = blA + blB
__global__ void prep_w1_kernel(const float* __restrict__ WlA, const float* __restrict__ WrA,
                               const float* __restrict__ WlB, const float* __restrict__ WrB,
                               const float* __restrict__ blA, const float* __restrict__ blB,
                               __bf16* __restrict__ W1, float* __restrict__ b1) {
  int idx = blockIdx.x * blockDim.x + threadIdx.x;
  if (idx >= H_HID * D_IN) return;
  int h = idx >> 7;
  int d = idx & (D_IN - 1);
  size_t base = (size_t)h * K_CAT;
  W1[base + d]          = f2bf(WlA[idx]);
  W1[base + D_IN + d]   = f2bf(WrA[idx] + WrB[idx]);
  W1[base + 2*D_IN + d] = f2bf(WlB[idx]);
  if (d == 0) b1[h] = blA[h] + blB[h];
}

__global__ void conv_bf16_kernel(const float* __restrict__ in, __bf16* __restrict__ out, int n) {
  int i = blockIdx.x * blockDim.x + threadIdx.x;
  if (i < n) out[i] = f2bf(in[i]);
}

// ---------------------------------------------------------------------------
// Fused:  hid = bf16(Xcat @ W1^T + b1)  (in LDS)
//         y   = hid @ Wp^T + bp
//         out = LayerNorm(y) * g + beta
// 32 rows / block, 8 waves. Stage1: each wave owns 32 hidden cols (2 N-tiles)
// x 2 M-tiles -> 4 accumulators, 12 K-steps of 32. Stage2: each wave owns 16
// of 128 output cols x 2 M-tiles, 8 K-steps from LDS.
// M must be a multiple of 32 (40000 and 200000 both are).
// ---------------------------------------------------------------------------
__launch_bounds__(256)
__global__ void fused_gemm_ln_kernel(const __bf16* __restrict__ Xcat,  // (M,384)
                                     const __bf16* __restrict__ W1,    // (256,384) row-major
                                     const float*  __restrict__ b1,    // (256)
                                     const __bf16* __restrict__ Wp,    // (128,256) row-major
                                     const float*  __restrict__ bp,    // (128)
                                     const float*  __restrict__ g,
                                     const float*  __restrict__ beta,
                                     float* __restrict__ out) {        // (M,128)
  __shared__ __align__(16) __bf16 hid[MT][H_HID + 8];   // +8 bf16 pad: kills bank conflicts
  __shared__ __align__(16) float  sout[MT][D_IN + 1];

  const int lane    = threadIdx.x & 31;
  const int wave    = threadIdx.x >> 5;
  const int lr      = lane & 15;
  const int halfsel = lane >> 4;              // 0: K-low half lanes, 1: K-high half lanes
  const int mofs    = halfsel * 8;            // C/D fragment: rows v (lanes 0-15) / v+8
  const int row_base = blockIdx.x * MT;

  // ---------------- Stage 1 ----------------
  v8f acc[2][2];
  #pragma unroll
  for (int mt = 0; mt < 2; ++mt)
    #pragma unroll
    for (int nt = 0; nt < 2; ++nt)
      acc[mt][nt] = (v8f)0.0f;

  #pragma unroll 4
  for (int kt = 0; kt < K_CAT / 32; ++kt) {
    const int k0 = kt * 32;
    // A fragments (16x32 bf16): lane<16 -> K[k0..k0+15]; lane>=16 -> K[k0+16..k0+31]
    v16bf afr[2];
    #pragma unroll
    for (int mt = 0; mt < 2; ++mt) {
      const int r  = row_base + mt * 16 + lr;
      const int kk = k0 + halfsel * 16;
      const __bf16* p = Xcat + (size_t)r * K_CAT + kk;
      afr[mt] = frag_pair(p, p + 8);
    }
    // B fragments (32x16 bf16): lane = column, 16 contiguous K per lane
    v16bf bfr[2];
    #pragma unroll
    for (int nt = 0; nt < 2; ++nt) {
      const int h  = wave * 32 + nt * 16 + lr;
      const int kb = k0 + halfsel * 16;
      bfr[nt] = *(const v16bf*)(W1 + (size_t)h * K_CAT + kb);
    }
    #pragma unroll
    for (int mt = 0; mt < 2; ++mt)
      #pragma unroll
      for (int nt = 0; nt < 2; ++nt)
        acc[mt][nt] = __builtin_amdgcn_wmma_f32_16x16x32_bf16(
            false, afr[mt], false, bfr[nt], (short)0, acc[mt][nt], false, false);
  }

  // bias + convert, stash hidden tile in LDS as bf16
  #pragma unroll
  for (int nt = 0; nt < 2; ++nt) {
    const int n = wave * 32 + nt * 16 + lr;
    const float bia = b1[n];
    #pragma unroll
    for (int mt = 0; mt < 2; ++mt)
      #pragma unroll
      for (int v = 0; v < 8; ++v)
        hid[mt * 16 + mofs + v][n] = f2bf(acc[mt][nt][v] + bia);
  }
  __syncthreads();

  // ---------------- Stage 2 ----------------
  v8f acc2[2];
  acc2[0] = (v8f)0.0f;
  acc2[1] = (v8f)0.0f;

  #pragma unroll
  for (int kt = 0; kt < H_HID / 32; ++kt) {
    const int k0 = kt * 32;
    v16bf afr[2];
    #pragma unroll
    for (int mt = 0; mt < 2; ++mt) {
      const int r  = mt * 16 + lr;
      const int kk = k0 + halfsel * 16;
      afr[mt] = frag_pair(&hid[r][kk], &hid[r][kk + 8]);
    }
    const int col = wave * 16 + lr;
    const int kb  = k0 + halfsel * 16;
    v16bf bfr = *(const v16bf*)(Wp + (size_t)col * H_HID + kb);
    #pragma unroll
    for (int mt = 0; mt < 2; ++mt)
      acc2[mt] = __builtin_amdgcn_wmma_f32_16x16x32_bf16(
          false, afr[mt], false, bfr, (short)0, acc2[mt], false, false);
  }

  {
    const int n = wave * 16 + lr;
    const float bia = bp[n];
    #pragma unroll
    for (int mt = 0; mt < 2; ++mt)
      #pragma unroll
      for (int v = 0; v < 8; ++v)
        sout[mt * 16 + mofs + v][n] = acc2[mt][v] + bia;
  }
  __syncthreads();

  // ---------------- LayerNorm ----------------
  const int row = threadIdx.x >> 3;   // 32 rows x 8 lanes
  const int sub = threadIdx.x & 7;
  float s = 0.0f, s2 = 0.0f;
  #pragma unroll
  for (int j = sub; j < D_IN; j += 8) {
    float v = sout[row][j];
    s += v; s2 += v * v;
  }
  #pragma unroll
  for (int m = 4; m; m >>= 1) {
    s  += __shfl_xor(s,  m, 8);
    s2 += __shfl_xor(s2, m, 8);
  }
  const float mu   = s * (1.0f / D_IN);
  const float var  = s2 * (1.0f / D_IN) - mu * mu;
  const float rstd = rsqrtf(var + LN_EPSF);
  float* orow = out + (size_t)(row_base + row) * D_IN;
  #pragma unroll
  for (int j = sub; j < D_IN; j += 8)
    orow[j] = (sout[row][j] - mu) * rstd * g[j] + beta[j];
}

// ---------------------------------------------------------------------------
extern "C" void kernel_launch(void* const* d_in, const int* in_sizes, int n_in,
                              void* d_out, int out_size, void* d_ws, size_t ws_size,
                              hipStream_t stream) {
  (void)n_in; (void)out_size; (void)ws_size;
  const float* z1 = (const float*)d_in[0];
  const float* z2 = (const float*)d_in[1];
  const int* e_pipe = (const int*)d_in[2];
  const int* e_surf = (const int*)d_in[3];
  const int* e_c2   = (const int*)d_in[4];
  const int* e_cf   = (const int*)d_in[5];
  const float* Wl_pipe = (const float*)d_in[6];
  const float* bl_pipe = (const float*)d_in[7];
  const float* Wr_pipe = (const float*)d_in[8];
  const float* Wl_surf = (const float*)d_in[9];
  const float* bl_surf = (const float*)d_in[10];
  const float* Wr_surf = (const float*)d_in[11];
  const float* Wl_c2   = (const float*)d_in[12];
  const float* bl_c2   = (const float*)d_in[13];
  const float* Wr_c2   = (const float*)d_in[14];
  const float* Wl_cf   = (const float*)d_in[15];
  const float* bl_cf   = (const float*)d_in[16];
  const float* Wr_cf   = (const float*)d_in[17];
  const float* W_p1 = (const float*)d_in[18];
  const float* b_p1 = (const float*)d_in[19];
  const float* W_p2 = (const float*)d_in[20];
  const float* b_p2 = (const float*)d_in[21];
  const float* g1    = (const float*)d_in[22];
  const float* beta1 = (const float*)d_in[23];
  const float* g2    = (const float*)d_in[24];
  const float* beta2 = (const float*)d_in[25];

  const int N1 = 20000, N2 = 100000;
  const int E_PIPE = in_sizes[2] / 2;
  const int E_SURF = in_sizes[3] / 2;
  const int E_C2   = in_sizes[4] / 2;
  const int E_CF   = in_sizes[5] / 2;

  // ---- workspace carve-up (256B aligned) ----
  char* ws = (char*)d_ws;
  size_t off = 0;
  auto carve = [&](size_t bytes) -> void* {
    void* p = ws + off;
    off = (off + bytes + 255) & ~(size_t)255;
    return p;
  };
  float* agg1A = (float*)carve((size_t)2 * N1 * D_IN * 4);   // pipe -> 1d
  float* agg1B = (float*)carve((size_t)2 * N1 * D_IN * 4);   // cf   -> 1d
  float* agg2A = (float*)carve((size_t)2 * N2 * D_IN * 4);   // surf -> 2d
  float* agg2B = (float*)carve((size_t)2 * N2 * D_IN * 4);   // c2   -> 2d
  float* cnt1A = (float*)carve((size_t)N1 * 4);
  float* cnt1B = (float*)carve((size_t)N1 * 4);
  float* cnt2A = (float*)carve((size_t)N2 * 4);
  float* cnt2B = (float*)carve((size_t)N2 * 4);
  size_t zero_bytes = off;                                    // everything above needs zeroing
  __bf16* xcat1 = (__bf16*)carve((size_t)2 * N1 * K_CAT * 2);
  __bf16* xcat2 = (__bf16*)carve((size_t)2 * N2 * K_CAT * 2);
  __bf16* W1_1d = (__bf16*)carve((size_t)H_HID * K_CAT * 2);
  __bf16* W1_2d = (__bf16*)carve((size_t)H_HID * K_CAT * 2);
  float*  b1_1d = (float*)carve((size_t)H_HID * 4);
  float*  b1_2d = (float*)carve((size_t)H_HID * 4);
  __bf16* Wp1   = (__bf16*)carve((size_t)D_IN * H_HID * 2);
  __bf16* Wp2   = (__bf16*)carve((size_t)D_IN * H_HID * 2);

  // 1) zero accumulators + counts
  zero_f32_kernel<<<2048, 256, 0, stream>>>((float*)ws, (long)(zero_bytes / 4));

  // 2) scatter-add (mean numerators + counts)
  auto sg = [](int E) { return (E * D_IN + 255) / 256; };
  scatter_accum_kernel<<<sg(E_PIPE), 256, 0, stream>>>(z1, e_pipe, agg1A, cnt1A, E_PIPE, N1, N1);
  scatter_accum_kernel<<<sg(E_CF),   256, 0, stream>>>(z2, e_cf,   agg1B, cnt1B, E_CF,   N2, N1);
  scatter_accum_kernel<<<sg(E_SURF), 256, 0, stream>>>(z2, e_surf, agg2A, cnt2A, E_SURF, N2, N2);
  scatter_accum_kernel<<<sg(E_C2),   256, 0, stream>>>(z1, e_c2,   agg2B, cnt2B, E_C2,   N1, N2);

  // 3) normalize + pack bf16 concatenated activations
  pack_xcat_kernel<<<(2 * N1 * D_IN + 255) / 256, 256, 0, stream>>>(z1, agg1A, agg1B, cnt1A, cnt1B, xcat1, N1);
  pack_xcat_kernel<<<(2 * N2 * D_IN + 255) / 256, 256, 0, stream>>>(z2, agg2A, agg2B, cnt2A, cnt2B, xcat2, N2);

  // 4) weight prep: W1 = [WlA | WrA+WrB | WlB], b1 = blA+blB; Wp bf16
  prep_w1_kernel<<<(H_HID * D_IN + 255) / 256, 256, 0, stream>>>(
      Wl_pipe, Wr_pipe, Wl_cf, Wr_cf, bl_pipe, bl_cf, W1_1d, b1_1d);
  prep_w1_kernel<<<(H_HID * D_IN + 255) / 256, 256, 0, stream>>>(
      Wl_surf, Wr_surf, Wl_c2, Wr_c2, bl_surf, bl_c2, W1_2d, b1_2d);
  conv_bf16_kernel<<<(D_IN * H_HID + 255) / 256, 256, 0, stream>>>(W_p1, Wp1, D_IN * H_HID);
  conv_bf16_kernel<<<(D_IN * H_HID + 255) / 256, 256, 0, stream>>>(W_p2, Wp2, D_IN * H_HID);

  // 5) fused WMMA GEMM->GEMM->LN  (M multiples of 32: 40000, 200000)
  float* out1 = (float*)d_out;
  float* out2 = out1 + (size_t)2 * N1 * D_IN;
  fused_gemm_ln_kernel<<<(2 * N1) / MT, 256, 0, stream>>>(
      xcat1, W1_1d, b1_1d, Wp1, b_p1, g1, beta1, out1);
  fused_gemm_ln_kernel<<<(2 * N2) / MT, 256, 0, stream>>>(
      xcat2, W1_2d, b1_2d, Wp2, b_p2, g2, beta2, out2);
}
